// NewSplitRTrainer_85461259256172
// MI455X (gfx1250) — matrix operand order repaired and checked by reading, not compiled, not run on hardware
//
#include <hip/hip_runtime.h>

// ---------------------------------------------------------------------------
// Problem constants (from reference)
// ---------------------------------------------------------------------------
constexpr int T_ = 4, C_ = 2, S_ = 2, U_ = C_ * S_;        // U = 4
constexpr int HD_ = 1024, PD_ = 512, SD_ = 256;
constexpr int BZ_ = 1024, L_ = 4096, STEPS_ = 8;

typedef unsigned short u16;
typedef unsigned long long u64;
typedef __attribute__((ext_vector_type(16))) u16     v16u;
typedef __attribute__((ext_vector_type(16))) __bf16  v16bf;
typedef __attribute__((ext_vector_type(8)))  float   v8f;

// ---------------------------------------------------------------------------
// Helpers
// ---------------------------------------------------------------------------
__device__ inline u16 f2bf(float x) {
    unsigned u = __float_as_uint(x);
    unsigned r = u + 0x7FFFu + ((u >> 16) & 1u);   // round-to-nearest-even
    return (u16)(r >> 16);
}

__device__ inline u64 pack4bf(float4 f) {
    return (u64)f2bf(f.x) | ((u64)f2bf(f.y) << 16) |
           ((u64)f2bf(f.z) << 32) | ((u64)f2bf(f.w) << 48);
}

__device__ inline v16bf as_bf(v16u x) {
    union { v16u u; v16bf b; } c; c.u = x; return c.b;
}

// A-matrix (16x32 bf16) fragment: lanes 0-15 = rows 0-15 K{0..7,16..23},
// lanes 16-31 = rows 0-15 K{8..15,24..31}  (CDNA5 ISA 7.12.2)
__device__ inline v16u load_frag_A(const u16* base, int ld, int row0, int k0, int lane) {
    const u16* p = base + (long long)(row0 + (lane & 15)) * ld + k0 + ((lane >> 4) << 3);
    v16u f;
#pragma unroll
    for (int i = 0; i < 8; ++i) f[i] = p[i];
#pragma unroll
    for (int i = 0; i < 8; ++i) f[i + 8] = p[i + 16];
    return f;
}

// B-matrix (32x16 bf16) fragment from "rows are K-contiguous" storage:
// lane n holds column n; lanes 0-15 K{0..7,16..23}, lanes 16-31 K{8..15,24..31}.
__device__ inline v16u load_frag_B(const u16* base, int ld, int col0, int k0, int lane) {
    const u16* p = base + (long long)(col0 + (lane & 15)) * ld + k0 + ((lane >> 4) << 3);
    v16u f;
#pragma unroll
    for (int i = 0; i < 8; ++i) f[i] = p[i];
#pragma unroll
    for (int i = 0; i < 8; ++i) f[i + 8] = p[i + 16];
    return f;
}

__device__ inline v8f wmma_bf16(v16u a, v16u b, v8f c) {
    return __builtin_amdgcn_wmma_f32_16x16x32_bf16(
        false, as_bf(a), false, as_bf(b), (short)0, c, false, false);
}

// ---------------------------------------------------------------------------
// f32 -> bf16 conversion
// ---------------------------------------------------------------------------
__global__ void cvt_bf16_k(const float* __restrict__ x, u16* __restrict__ y, long long n) {
    long long i = (long long)blockIdx.x * blockDim.x + threadIdx.x;
    long long stride = (long long)gridDim.x * blockDim.x;
    for (; i < n; i += stride) y[i] = f2bf(x[i]);
}

// ---------------------------------------------------------------------------
// Generic batched WMMA GEMM:  C[z] = A[z] (MxK) * B[bb] (KxN), bf16 in, f32 acc.
// Block tile 128x64: 8 waves (4 in M x 2 in N), wave tile 32x32 (4 WMMAs per
// K-chunk of 32).  Register double-buffering: next chunk's global loads are
// issued while current chunk's WMMAs execute.  Staging is 8-byte vectorized.
// Optional gathered f32 A (rowidx), batch-derived K-offset into A and
// N-offset / batch index into B.
// ---------------------------------------------------------------------------
template <bool OUT_BF16>
__global__ __launch_bounds__(256) void gemm_bf16_k(
    const u16*   __restrict__ A16,
    const float* __restrict__ A32,
    const int*   __restrict__ rowidx,
    const u16*   __restrict__ B,
    void*        __restrict__ Cv,
    int M, int N, int K,
    int lda, int ldb, int ldc,
    long long a_bstride, long long c_bstride, long long b_bstride,
    int acol_mode,     // 0: 0     1: (z%C)*PD
    int b_batch_mode,  // 0: z     1: (z/U)*C + (z%U)/S     2: 0
    int bcol_mode)     // 0: 0     1: ((z%U)/S)*PD          2: ((z%U)%S)*SD
{
    __shared__ u16 As[128 * 40];   // 10.0 KB
    __shared__ u16 Bt[64 * 40];    //  5.0 KB (B tile transposed: [n][k])

    const int tid  = threadIdx.x;
    const int lane = tid & 31;
    const int wave = tid >> 5;
    const int wm   = wave & 3;     // 4 waves in M
    const int wn   = wave >> 2;    // 2 waves in N
    const int z    = blockIdx.z;

    const int acol0 = (acol_mode == 1) ? (z % C_) * PD_ : 0;
    int bb = z;
    if (b_batch_mode == 1)      bb = (z / U_) * C_ + (z % U_) / S_;
    else if (b_batch_mode == 2) bb = 0;
    int bcol0 = 0;
    if (bcol_mode == 1)      bcol0 = ((z % U_) / S_) * PD_;
    else if (bcol_mode == 2) bcol0 = ((z % U_) % S_) * SD_;

    const u16*   Ab   = A16 ? A16 + (long long)z * a_bstride : nullptr;
    const float* A32b = A32 ? A32 + (long long)z * a_bstride : nullptr;
    const int*   ridx = rowidx ? rowidx + (long long)z * M : nullptr;
    const u16*   Bb   = B + (long long)bb * b_bstride;

    const int Mbase = blockIdx.y * 128;
    const int Nbase = blockIdx.x * 64;

    // staging registers (double buffer lives in regs, single LDS buffer)
    u64 aR[4];   // A: 128x32 u16 = 1024 u64 slots, 4/thread
    u64 bR[2];   // B:  64x32 u16 =  512 u64 slots, 2/thread

    auto fetch = [&](int kc) {
#pragma unroll
        for (int i = 0; i < 4; ++i) {
            int s = tid + i * 256;
            int r = s >> 3, cq = (s & 7) << 2;       // row, col(quad)
            long long grow = Mbase + r;
            if (ridx) grow = ridx[Mbase + r];
            long long off = grow * (long long)lda + acol0 + kc + cq;
            aR[i] = A32b ? pack4bf(*(const float4*)(A32b + off))
                         : *(const u64*)(Ab + off);
        }
#pragma unroll
        for (int i = 0; i < 2; ++i) {
            int s = tid + i * 256;
            int kk = s >> 4, n0 = (s & 15) << 2;     // K row, N col(quad)
            bR[i] = *(const u64*)(Bb + (long long)(kc + kk) * ldb + bcol0 + Nbase + n0);
        }
    };

    v8f acc00 = {}, acc01 = {}, acc10 = {}, acc11 = {};

    fetch(0);
    for (int k0 = 0; k0 < K; k0 += 32) {
        // regs -> LDS
#pragma unroll
        for (int i = 0; i < 4; ++i) {
            int s = tid + i * 256;
            int r = s >> 3, cq = (s & 7) << 2;
            *(u64*)(&As[r * 40 + cq]) = aR[i];
        }
#pragma unroll
        for (int i = 0; i < 2; ++i) {
            int s = tid + i * 256;
            int kk = s >> 4, n0 = (s & 15) << 2;
            u64 v = bR[i];
#pragma unroll
            for (int j = 0; j < 4; ++j)
                Bt[(n0 + j) * 40 + kk] = (u16)(v >> (16 * j));
        }
        __syncthreads();

        if (k0 + 32 < K) fetch(k0 + 32);   // overlap next chunk's loads

        v16u a0 = load_frag_A(As, 40, wm * 32,      0, lane);
        v16u a1 = load_frag_A(As, 40, wm * 32 + 16, 0, lane);
        v16u b0 = load_frag_B(Bt, 40, wn * 32,      0, lane);
        v16u b1 = load_frag_B(Bt, 40, wn * 32 + 16, 0, lane);
        acc00 = wmma_bf16(a0, b0, acc00);
        acc01 = wmma_bf16(a0, b1, acc01);
        acc10 = wmma_bf16(a1, b0, acc10);
        acc11 = wmma_bf16(a1, b1, acc11);
        __syncthreads();
    }

    // epilogue: C layout VGPR r -> row r (lanes 0-15) / r+8 (lanes 16-31)
    const int n0c = Nbase + wn * 32 + (lane & 15);
    const int m0  = Mbase + wm * 32 + ((lane >> 4) << 3);
    if constexpr (OUT_BF16) {
        u16* Cb = (u16*)Cv + (long long)z * c_bstride;
#pragma unroll
        for (int r = 0; r < 8; ++r) {
            Cb[(long long)(m0 + r)      * ldc + n0c]      = f2bf(acc00[r]);
            Cb[(long long)(m0 + r)      * ldc + n0c + 16] = f2bf(acc01[r]);
            Cb[(long long)(m0 + 16 + r) * ldc + n0c]      = f2bf(acc10[r]);
            Cb[(long long)(m0 + 16 + r) * ldc + n0c + 16] = f2bf(acc11[r]);
        }
    } else {
        float* Cb = (float*)Cv + (long long)z * c_bstride;
#pragma unroll
        for (int r = 0; r < 8; ++r) {
            Cb[(long long)(m0 + r)      * ldc + n0c]      = acc00[r];
            Cb[(long long)(m0 + r)      * ldc + n0c + 16] = acc01[r];
            Cb[(long long)(m0 + 16 + r) * ldc + n0c]      = acc10[r];
            Cb[(long long)(m0 + 16 + r) * ldc + n0c + 16] = acc11[r];
        }
    }
}

// ---------------------------------------------------------------------------
// Row normalize (rows of SD=256): y = x / max(||x||, clipmin), bf16 out
// ---------------------------------------------------------------------------
__global__ __launch_bounds__(256) void rownorm_k(const float* __restrict__ x,
                                                 u16* __restrict__ y,
                                                 float clipmin) {
    __shared__ float red[256];
    long long row = blockIdx.x;
    float v = x[row * SD_ + threadIdx.x];
    red[threadIdx.x] = v * v;
    __syncthreads();
    for (int s = 128; s > 0; s >>= 1) {
        if (threadIdx.x < s) red[threadIdx.x] += red[threadIdx.x + s];
        __syncthreads();
    }
    float sc = 1.0f / fmaxf(sqrtf(red[0]), clipmin);
    y[row * SD_ + threadIdx.x] = f2bf(v * sc);
}

// ---------------------------------------------------------------------------
// Fused sim = qn @ kn^T with running per-row max/argmax.
// grid = (BZ/16, T*U). Block = 8 waves; each wave sweeps 1/8 of L's N tiles,
// two tiles per iteration (overlapped b128 loads between the two WMMA chains).
// Per-lane running (val,idx) across all tiles; the cross-lane __shfl_xor
// reduction happens ONCE after the sweep instead of per tile.
// ---------------------------------------------------------------------------
__global__ __launch_bounds__(256) void sim_argmax_k(
    const u16* __restrict__ qn, const u16* __restrict__ kn,
    float* __restrict__ best_val, int* __restrict__ best_idx, int step)
{
    const int tid = threadIdx.x, lane = tid & 31, wave = tid >> 5;
    const int tu = blockIdx.y;
    const int t = tu / U_, u = tu % U_, c = u / S_, s = u % S_;
    const int Mbase = blockIdx.x * 16;

    const u16* Ab = qn + ((long long)(t * C_ + c) * BZ_) * PD_ + s * SD_;
    const u16* Bb = kn + ((long long)(t * C_ + c) * L_) * PD_ + s * SD_;

    v16u af[8];
#pragma unroll
    for (int kk = 0; kk < 8; ++kk) af[kk] = load_frag_A(Ab, PD_, Mbase, kk * 32, lane);

    float bv[8]; int bi[8];
#pragma unroll
    for (int r = 0; r < 8; ++r) { bv[r] = -1e30f; bi[r] = 0; }

    const int mycol = lane & 15;
    for (int nt = wave; nt < L_ / 16; nt += 16) {
        int Nb0 = nt * 16;
        int Nb1 = (nt + 8) * 16;
        if (nt + 16 < L_ / 16)   // prefetch next iteration's rows (gfx1250 path)
            __builtin_prefetch(Bb + (long long)((nt + 16) * 16 + mycol) * PD_ +
                               ((lane >> 4) << 4), 0, 0);
        v8f acc0 = {}, acc1 = {};
#pragma unroll
        for (int kk = 0; kk < 8; ++kk) {
            v16u b0 = load_frag_B(Bb, PD_, Nb0, kk * 32, lane);
            v16u b1 = load_frag_B(Bb, PD_, Nb1, kk * 32, lane);
            acc0 = wmma_bf16(af[kk], b0, acc0);
            acc1 = wmma_bf16(af[kk], b1, acc1);
        }
#pragma unroll
        for (int r = 0; r < 8; ++r) {
            bool g0 = acc0[r] > bv[r];
            bv[r] = g0 ? acc0[r] : bv[r];
            bi[r] = g0 ? Nb0 + mycol : bi[r];
            bool g1 = acc1[r] > bv[r];
            bv[r] = g1 ? acc1[r] : bv[r];
            bi[r] = g1 ? Nb1 + mycol : bi[r];
        }
    }

    // one cross-lane reduction within each 16-lane half (branchless)
#pragma unroll
    for (int r = 0; r < 8; ++r) {
        float v = bv[r]; int i = bi[r];
#pragma unroll
        for (int off = 1; off < 16; off <<= 1) {
            float ov = __shfl_xor(v, off, 32);
            int   oi = __shfl_xor(i, off, 32);
            bool g = ov > v;
            v = g ? ov : v;
            i = g ? oi : i;
        }
        bv[r] = v; bi[r] = i;
    }

    __shared__ float sv[8][16];
    __shared__ int   si[8][16];
    if ((lane & 15) == 0) {
        int half = lane >> 4;
#pragma unroll
        for (int r = 0; r < 8; ++r) {
            sv[wave][half * 8 + r] = bv[r];
            si[wave][half * 8 + r] = bi[r];
        }
    }
    __syncthreads();
    if (tid < 16) {
        float v = sv[0][tid]; int i = si[0][tid];
#pragma unroll
        for (int w = 1; w < 8; ++w) {
            bool g = sv[w][tid] > v;
            v = g ? sv[w][tid] : v;
            i = g ? si[w][tid] : i;
        }
        long long o = (long long)tu * BZ_ + Mbase + tid;
        if (v > best_val[o]) { best_val[o] = v; best_idx[o] = step * L_ + i; }
    }
}

// ---------------------------------------------------------------------------
// init best_val/-1, best_idx/0
// ---------------------------------------------------------------------------
__global__ void init_best_k(float* bv, int* bi, int n) {
    int i = blockIdx.x * blockDim.x + threadIdx.x;
    if (i < n) { bv[i] = -1.0f; bi[i] = 0; }
}

// ---------------------------------------------------------------------------
// loss: cos(q, ms) per (t,u,b); atomic-accumulate -SD/(HD*BZ) * cos
// ---------------------------------------------------------------------------
__global__ __launch_bounds__(256) void loss_k(const float* __restrict__ q,
                                              const float* __restrict__ ms,
                                              float* __restrict__ out) {
    long long rid = blockIdx.x;          // [T,U,BZ]
    int b = (int)(rid % BZ_);
    long long tu = rid / BZ_;
    int t = (int)(tu / U_), u = (int)(tu % U_), c = u / S_, s = u % S_;

    const float* qr = q + ((long long)(t * C_ + c) * BZ_ + b) * PD_ + s * SD_;
    const float* mr = ms + rid * SD_;

    int i = threadIdx.x;
    float qv = qr[i], mv = mr[i];
    __shared__ float rd[256], rq[256], rm[256];
    rd[i] = qv * mv; rq[i] = qv * qv; rm[i] = mv * mv;
    __syncthreads();
    for (int st = 128; st > 0; st >>= 1) {
        if (i < st) { rd[i] += rd[i + st]; rq[i] += rq[i + st]; rm[i] += rm[i + st]; }
        __syncthreads();
    }
    if (i == 0) {
        float cosv = rd[0] / (fmaxf(sqrtf(rq[0]), 1e-8f) * fmaxf(sqrtf(rm[0]), 1e-8f));
        atomicAdd(out, cosv * (-(float)SD_ / ((float)HD_ * (float)BZ_)));
    }
}

// ---------------------------------------------------------------------------
// Host launcher
// ---------------------------------------------------------------------------
extern "C" void kernel_launch(void* const* d_in, const int* in_sizes, int n_in,
                              void* d_out, int out_size, void* d_ws, size_t ws_size,
                              hipStream_t stream) {
    (void)in_sizes; (void)n_in; (void)out_size; (void)ws_size;

    const float* h    = (const float*)d_in[0];   // [BZ, HD]
    const float* keys = (const float*)d_in[1];   // [STEPS, L, HD]
    const float* pR   = (const float*)d_in[2];   // [HD, HD]
    const float* Rs   = (const float*)d_in[3];   // [T, C, PD, PD]
    float* out = (float*)d_out;

    // workspace carve (256B aligned)
    char* w = (char*)d_ws;
    auto alloc = [&](long long bytes) -> void* {
        void* p = (void*)w;
        w += (bytes + 255) & ~255ll;
        return p;
    };
    u16*   pR_bf = (u16*)  alloc((long long)HD_ * HD_ * 2);
    u16*   Rs_bf = (u16*)  alloc((long long)T_ * C_ * PD_ * PD_ * 2);
    u16*   h_bf  = (u16*)  alloc((long long)BZ_ * HD_ * 2);
    u16*   hr_bf = (u16*)  alloc((long long)BZ_ * HD_ * 2);
    float* q_f   = (float*)alloc((long long)T_ * C_ * BZ_ * PD_ * 4);
    u16*   qn_bf = (u16*)  alloc((long long)T_ * C_ * BZ_ * PD_ * 2);
    u16*   kb_bf = (u16*)  alloc((long long)L_ * HD_ * 2);
    u16*   kr_bf = (u16*)  alloc((long long)L_ * HD_ * 2);
    float* ks_f  = (float*)alloc((long long)T_ * C_ * L_ * PD_ * 4);
    u16*   kn_bf = (u16*)  alloc((long long)T_ * C_ * L_ * PD_ * 2);
    float* bv    = (float*)alloc((long long)T_ * U_ * BZ_ * 4);
    int*   bi    = (int*)  alloc((long long)T_ * U_ * BZ_ * 4);
    u16*   tmp_bf= (u16*)  alloc((long long)T_ * U_ * BZ_ * PD_ * 2);
    float* ms_f  = (float*)alloc((long long)T_ * U_ * BZ_ * SD_ * 4);

    hipMemsetAsync(d_out, 0, sizeof(float), stream);

    // bf16 conversions
    cvt_bf16_k<<<2048, 256, 0, stream>>>(pR, pR_bf, (long long)HD_ * HD_);
    cvt_bf16_k<<<2048, 256, 0, stream>>>(Rs, Rs_bf, (long long)T_ * C_ * PD_ * PD_);
    cvt_bf16_k<<<1024, 256, 0, stream>>>(h, h_bf, (long long)BZ_ * HD_);

    // hr = h @ pR  (bf16 out)
    gemm_bf16_k<true><<<dim3(HD_ / 64, BZ_ / 128, 1), 256, 0, stream>>>(
        h_bf, nullptr, nullptr, pR_bf, hr_bf,
        BZ_, HD_, HD_, HD_, HD_, HD_, 0, 0, 0, 0, 2, 0);

    // q[t,c] = hr[:, c*PD:] @ Rs[t,c]  (f32 out), batches z = t*C+c
    gemm_bf16_k<false><<<dim3(PD_ / 64, BZ_ / 128, T_ * C_), 256, 0, stream>>>(
        hr_bf, nullptr, nullptr, Rs_bf, q_f,
        BZ_, PD_, PD_, HD_, PD_, PD_,
        0, (long long)BZ_ * PD_, (long long)PD_ * PD_, 1, 0, 0);

    // qn = normalize rows of SD (clip 1e-8)
    rownorm_k<<<T_ * C_ * BZ_ * S_, 256, 0, stream>>>(q_f, qn_bf, 1e-8f);

    init_best_k<<<(T_ * U_ * BZ_ + 255) / 256, 256, 0, stream>>>(bv, bi, T_ * U_ * BZ_);

    for (int step = 0; step < STEPS_; ++step) {
        const float* kb = keys + (long long)step * L_ * HD_;
        cvt_bf16_k<<<4096, 256, 0, stream>>>(kb, kb_bf, (long long)L_ * HD_);

        // kr = kb @ pR (bf16 out)
        gemm_bf16_k<true><<<dim3(HD_ / 64, L_ / 128, 1), 256, 0, stream>>>(
            kb_bf, nullptr, nullptr, pR_bf, kr_bf,
            L_, HD_, HD_, HD_, HD_, HD_, 0, 0, 0, 0, 2, 0);

        // ks[t,c] = kr[:, c*PD:] @ Rs[t,c] (f32 out)
        gemm_bf16_k<false><<<dim3(PD_ / 64, L_ / 128, T_ * C_), 256, 0, stream>>>(
            kr_bf, nullptr, nullptr, Rs_bf, ks_f,
            L_, PD_, PD_, HD_, PD_, PD_,
            0, (long long)L_ * PD_, (long long)PD_ * PD_, 1, 0, 0);

        // kn = normalize rows of SD (clip 1e-3)
        rownorm_k<<<T_ * C_ * L_ * S_, 256, 0, stream>>>(ks_f, kn_bf, 1e-3f);

        // fused sim + running max/argmax
        sim_argmax_k<<<dim3(BZ_ / 16, T_ * U_), 256, 0, stream>>>(qn_bf, kn_bf, bv, bi, step);
    }

    // tmp[t,u] = keys[best_idx[t,u,:]] @ pR[:, c*PD:] (gather-A f32, bf16 out)
    gemm_bf16_k<true><<<dim3(PD_ / 64, BZ_ / 128, T_ * U_), 256, 0, stream>>>(
        nullptr, keys, bi, pR_bf, tmp_bf,
        BZ_, PD_, HD_, HD_, HD_, PD_,
        0, (long long)BZ_ * PD_, 0, 0, 2, 1);

    // ms[t,u] = tmp[t,u] @ Rs[t,c][:, s*SD:] (f32 out)
    gemm_bf16_k<false><<<dim3(SD_ / 64, BZ_ / 128, T_ * U_), 256, 0, stream>>>(
        tmp_bf, nullptr, nullptr, Rs_bf, ms_f,
        BZ_, SD_, PD_, PD_, PD_, SD_,
        (long long)BZ_ * PD_, (long long)BZ_ * SD_, (long long)PD_ * PD_, 0, 1, 2);

    // loss reduction
    loss_k<<<T_ * U_ * BZ_, 256, 0, stream>>>(q_f, ms_f, out);
}
